// CausalAttention_9259949490567
// MI455X (gfx1250) — matrix operand order repaired and checked by reading, not compiled, not run on hardware
//
#include <hip/hip_runtime.h>
#include <hip/hip_bf16.h>

// B=4, S=2048, D_IN=D_OUT=1024, M = B*S = 8192
typedef __attribute__((ext_vector_type(16))) __bf16 v16bf;
typedef __attribute__((ext_vector_type(8)))  __bf16 v8bf;
typedef __attribute__((ext_vector_type(8)))  float  v8f;

#define WMMA_BF16(a, b, c) \
  __builtin_amdgcn_wmma_f32_16x16x32_bf16(false, (a), false, (b), (short)0, (c), false, false)

// LDS byte offset of a generic pointer known to point into shared memory.
__device__ __forceinline__ unsigned lds_offset(const void* p) {
  return (unsigned)(unsigned long long)(__attribute__((address_space(3))) const void*)p;
}

// A-matrix fragment (16x32 bf16, ISA 7.12.2): lane L<16 holds row L, K = {0..7, 16..23};
// lane L+16 holds row L, K = {8..15, 24..31}. Two 16B loads.
__device__ __forceinline__ v16bf load_fragA(const __bf16* base, int row0, int ld, int k0, int lane) {
  const __bf16* p = base + (size_t)(row0 + (lane & 15)) * ld + k0 + ((lane >> 4) << 3);
  v8bf lo = *(const v8bf*)p;
  v8bf hi = *(const v8bf*)(p + 16);
  v16bf r;
#pragma unroll
  for (int i = 0; i < 8; ++i) { r[i] = lo[i]; r[i + 8] = hi[i]; }
  return r;
}

// B-matrix fragment (32x16 bf16): lane = output column (mod 16), lanes 0-15 hold K={0..7,16..23},
// lanes 16-31 hold K={8..15,24..31}. Source array is stored [n][k] row-major, so K is contiguous.
__device__ __forceinline__ v16bf load_fragB(const __bf16* base, int n0, int ld, int k0, int lane) {
  const __bf16* p = base + (size_t)(n0 + (lane & 15)) * ld + k0 + ((lane >> 4) << 3);
  v8bf lo = *(const v8bf*)p;
  v8bf hi = *(const v8bf*)(p + 16);
  v16bf r;
#pragma unroll
  for (int i = 0; i < 8; ++i) { r[i] = lo[i]; r[i + 8] = hi[i]; }
  return r;
}

// ---------------- Kernel 1: Q/K/V = x @ W, f32 in, bf16 out (Q pre-scaled by 1/32) --------
__global__ __launch_bounds__(256) void qkv_gemm_kernel(
    const float* __restrict__ x, const float* __restrict__ Wq,
    const float* __restrict__ Wk, const float* __restrict__ Wv,
    __bf16* __restrict__ Qb, __bf16* __restrict__ Kb, __bf16* __restrict__ Vb) {
  __shared__ __bf16 Xs[128 * 40];   // [row][k], k-stride padded 32->40
  __shared__ __bf16 Ws[128 * 40];   // [n][k] (transposed W tile)
  const int t = threadIdx.x, lane = t & 31, wave = t >> 5;
  const int mBase = blockIdx.y * 128, nBase = blockIdx.x * 128;
  const int wm = (wave & 1) * 64, wn = (wave >> 1) * 32;   // wave tile 64x32
  const float* W; __bf16* Out; float scale;
  if (blockIdx.z == 0)      { W = Wq; Out = Qb; scale = 0.03125f; }  // fold 1/sqrt(1024)
  else if (blockIdx.z == 1) { W = Wk; Out = Kb; scale = 1.0f; }
  else                      { W = Wv; Out = Vb; scale = 1.0f; }
  v8f acc[4][2] = {};
  const int xr = t >> 1, xc = (t & 1) * 16;       // X stage: 2 thr/row, 16 f32 each
  const int wk = t >> 3, wn0 = (t & 7) * 16;      // W stage: coalesced read, transposed store
  for (int kk0 = 0; kk0 < 1024; kk0 += 32) {
    {
      const float* g = x + (size_t)(mBase + xr) * 1024 + kk0 + xc;
      v8bf v0, v1;
#pragma unroll
      for (int i = 0; i < 8; ++i) { v0[i] = (__bf16)g[i]; v1[i] = (__bf16)g[8 + i]; }
      *(v8bf*)(Xs + xr * 40 + xc) = v0;
      *(v8bf*)(Xs + xr * 40 + xc + 8) = v1;
    }
    {
      const float* g = W + (size_t)(kk0 + wk) * 1024 + nBase + wn0;
#pragma unroll
      for (int i = 0; i < 16; ++i) Ws[(wn0 + i) * 40 + wk] = (__bf16)g[i];
    }
    __syncthreads();
    v16bf a[4], bb[2];
#pragma unroll
    for (int ms = 0; ms < 4; ++ms) a[ms] = load_fragA(Xs, wm + ms * 16, 40, 0, lane);
#pragma unroll
    for (int ns = 0; ns < 2; ++ns) bb[ns] = load_fragB(Ws, wn + ns * 16, 40, 0, lane);
#pragma unroll
    for (int ms = 0; ms < 4; ++ms)
#pragma unroll
      for (int ns = 0; ns < 2; ++ns)
        acc[ms][ns] = WMMA_BF16(a[ms], bb[ns], acc[ms][ns]);
    __syncthreads();
  }
  const int rsub = (lane >> 4) << 3;   // C/D layout: VGPR r -> row r (lanes 0-15) / r+8 (16-31)
#pragma unroll
  for (int ms = 0; ms < 4; ++ms)
#pragma unroll
    for (int ns = 0; ns < 2; ++ns) {
      const int col = nBase + wn + ns * 16 + (lane & 15);
#pragma unroll
      for (int r = 0; r < 8; ++r) {
        const int row = mBase + wm + ms * 16 + rsub + r;
        Out[(size_t)row * 1024 + col] = (__bf16)(acc[ms][ns][r] * scale);
      }
    }
}

// ---------------- Kernel 2: causal scores + softmax stats (deferred 1/l) ------------------
// Block = 64 query rows (4 waves x 16). Each wave covers 64 keys per step so one Q
// A-fragment feeds 4 WMMAs. Q/K fragments read straight from L2-resident bf16.
__global__ __launch_bounds__(128) void attn_softmax_kernel(
    const __bf16* __restrict__ Qb, const __bf16* __restrict__ Kb,
    __bf16* __restrict__ Sraw, float* __restrict__ linv) {
  const int b = blockIdx.y;
  const int rb = blockIdx.x * 64;
  const int lane = threadIdx.x & 31, wave = threadIdx.x >> 5;
  const int rowBase = rb + wave * 16;
  const int rsub = (lane >> 4) << 3;
  const __bf16* Qrow = Qb + ((size_t)b * 2048 + rowBase) * 1024;
  float mrun[8];
#pragma unroll
  for (int r = 0; r < 8; ++r) mrun[r] = -__builtin_inff();
  const int ntiles = blockIdx.x + 1;   // 64-wide key tiles; causal bound keys < rb+64
  for (int j = 0; j < ntiles; ++j) {
    const __bf16* Krow = Kb + ((size_t)b * 2048 + j * 64) * 1024;
    v8f s[4] = {};
#pragma unroll 4
    for (int d = 0; d < 1024; d += 32) {
      v16bf aq = load_fragA(Qrow, 0, 1024, d, lane);
#pragma unroll
      for (int g = 0; g < 4; ++g) {
        v16bf kf = load_fragB(Krow, g * 16, 1024, d, lane);
        s[g] = WMMA_BF16(aq, kf, s[g]);
      }
    }
    const int c0 = j * 64 + (lane & 15);
    float tmax[8];
#pragma unroll
    for (int r = 0; r < 8; ++r) {
      const int row = rowBase + rsub + r;
      const size_t base = ((size_t)b * 2048 + row) * 2048;
      float mx = -__builtin_inff();
#pragma unroll
      for (int g = 0; g < 4; ++g) {
        const int c = c0 + g * 16;
        const float v = (c <= row) ? s[g][r] : -__builtin_inff();
        Sraw[base + c] = (__bf16)v;
        mx = fmaxf(mx, v);
      }
      tmax[r] = mx;
    }
#pragma unroll
    for (int m = 8; m >= 1; m >>= 1)
#pragma unroll
      for (int r = 0; r < 8; ++r)
        tmax[r] = fmaxf(tmax[r], __shfl_xor(tmax[r], m, 32));
#pragma unroll
    for (int r = 0; r < 8; ++r) mrun[r] = fmaxf(mrun[r], tmax[r]);
  }
  // Pass 2: in-place exp(s - m), accumulate row sums (block-local region, no cross-block race)
  float lsum[8] = {};
  const int ncols = ntiles * 64;
  for (int c = (lane & 15); c < ncols; c += 16) {
#pragma unroll
    for (int r = 0; r < 8; ++r) {
      const int row = rowBase + rsub + r;
      const size_t idx = ((size_t)b * 2048 + row) * 2048 + c;
      const float e = __expf((float)Sraw[idx] - mrun[r]);   // exp(-inf)=0 handles mask
      lsum[r] += e;
      Sraw[idx] = (__bf16)e;
    }
  }
#pragma unroll
  for (int m = 8; m >= 1; m >>= 1)
#pragma unroll
    for (int r = 0; r < 8; ++r) lsum[r] += __shfl_xor(lsum[r], m, 32);
  if ((lane & 15) == 0) {
#pragma unroll
    for (int r = 0; r < 8; ++r)
      linv[b * 2048 + rowBase + rsub + r] = 1.0f / lsum[r];
  }
}

// ---------------- Kernel 3: O = (P @ V) * diag(1/l), skipping masked key tiles ------------
__global__ __launch_bounds__(256) void attn_out_kernel(
    const __bf16* __restrict__ P, const __bf16* __restrict__ Vb,
    const float* __restrict__ linv, float* __restrict__ out) {
  __shared__ __bf16 Ps[64 * 40];    // [row][key] tile 64x32 (padded; async-filled contiguous 8s)
  __shared__ __bf16 Vt[128 * 40];   // [n][key]  tile 32x128 transposed
  const int b = blockIdx.z, rb = blockIdx.y * 64, nBase = blockIdx.x * 128;
  const int t = threadIdx.x, lane = t & 31, wave = t >> 5;
  const int wm = (wave & 3) * 16, wn = (wave >> 2) * 64;   // 4x2 wave grid over 64x128
  v8f acc[4] = {};
  const int ntiles = (rb + 64) >> 5;   // same causal bound kernel 2 wrote (32-key tiles)
  const int pr = t >> 2, pc = (t & 3) * 8;
  const int vk = t >> 3, vn = (t & 7) * 16;
  const __bf16* Pbase = P + ((size_t)b * 2048 + rb) * 2048;         // block-uniform -> SGPRs
  const unsigned pLds = lds_offset(Ps + pr * 40 + pc);              // LDS byte address
  for (int j = 0; j < ntiles; ++j) {
    // Async global->LDS copy of the P tile: 16B per lane, GVS mode (SGPR base + VGPR offset),
    // tracked by ASYNCcnt (ISA 10.x / async_tensor doc §4).
    {
      const unsigned voff = (unsigned)((pr * 2048 + j * 32 + pc) * 2);
      asm volatile("global_load_async_to_lds_b128 %0, %1, %2"
                   :: "v"(pLds), "v"(voff), "s"(Pbase) : "memory");
    }
    {
      const __bf16* src = Vb + ((size_t)b * 2048 + j * 32 + vk) * 1024 + nBase + vn;
#pragma unroll
      for (int i = 0; i < 16; ++i) Vt[(vn + i) * 40 + vk] = src[i];
    }
    asm volatile("s_wait_asynccnt 0" ::: "memory");
    __syncthreads();
    v16bf a = load_fragA(Ps, wm, 40, 0, lane);
#pragma unroll
    for (int ns = 0; ns < 4; ++ns) {
      v16bf bb = load_fragB(Vt, wn + ns * 16, 40, 0, lane);
      acc[ns] = WMMA_BF16(a, bb, acc[ns]);
    }
    __syncthreads();
  }
  const int rsub = (lane >> 4) << 3;
#pragma unroll
  for (int r = 0; r < 8; ++r) {
    const int row = rb + wm + rsub + r;
    const float inv = linv[b * 2048 + row];
#pragma unroll
    for (int ns = 0; ns < 4; ++ns)
      out[((size_t)b * 2048 + row) * 1024 + nBase + wn + ns * 16 + (lane & 15)] =
          acc[ns][r] * inv;
  }
}

extern "C" void kernel_launch(void* const* d_in, const int* in_sizes, int n_in,
                              void* d_out, int out_size, void* d_ws, size_t ws_size,
                              hipStream_t stream) {
  (void)in_sizes; (void)n_in; (void)out_size; (void)ws_size;
  const float* x  = (const float*)d_in[0];
  const float* Wq = (const float*)d_in[1];
  const float* Wk = (const float*)d_in[2];
  const float* Wv = (const float*)d_in[3];
  // Workspace layout: Qb/Kb/Vb bf16 [8192,1024], Sraw bf16 [4,2048,2048], linv f32 [4,2048]
  __bf16* Qb = (__bf16*)d_ws;
  __bf16* Kb = Qb + (size_t)8192 * 1024;
  __bf16* Vb = Kb + (size_t)8192 * 1024;
  __bf16* Sr = Vb + (size_t)8192 * 1024;
  float* linv = (float*)(Sr + (size_t)4 * 2048 * 2048);

  qkv_gemm_kernel<<<dim3(8, 64, 3), 256, 0, stream>>>(x, Wq, Wk, Wv, Qb, Kb, Vb);
  attn_softmax_kernel<<<dim3(32, 4), 128, 0, stream>>>(Qb, Kb, Sr, linv);
  attn_out_kernel<<<dim3(8, 32, 4), 256, 0, stream>>>(Sr, Vb, linv, (float*)d_out);
}
